// WindowCrossAttention_37572373905978
// MI455X (gfx1250) — compile-verified, matrix operand order checked by reading, not run on hardware
//
#include <hip/hip_runtime.h>
#include <hip/hip_bf16.h>

// ---------------- problem constants ----------------
constexpr int kDIM = 1024;
constexpr int kH   = 16;
constexpr int kHD  = 64;
constexpr int kL   = 2048;
constexpr int kB   = 2;
constexpr int kBL  = kB * kL;          // 4096 rows for the projections
constexpr float kSCALE = 0.125f;       // 64^-0.5
constexpr float kEPS   = 1e-6f;

typedef __attribute__((ext_vector_type(16))) __bf16 v16bf;
typedef __attribute__((ext_vector_type(8)))  __bf16 v8bf;
typedef __attribute__((ext_vector_type(4)))  __bf16 v4bf;
typedef __attribute__((ext_vector_type(8)))  float  v8f;

// ---------------- fragment helpers ----------------
__device__ __forceinline__ v8f wmma_bf16(v16bf a, v16bf b, v8f c) {
  return __builtin_amdgcn_wmma_f32_16x16x32_bf16(false, a, false, b,
                                                 (short)0, c, false, false);
}

// A/B fragment from a K-contiguous bf16 row. Lane layout per ISA:
// elements 0..7  = row[k0 + half*8 .. +7]
// elements 8..15 = row[k0 + 16 + half*8 .. +7]
__device__ __forceinline__ v16bf load_frag_bf16(const __bf16* row, int k0, int half) {
  v8bf a = *(const v8bf*)(row + k0 + half * 8);
  v8bf b = *(const v8bf*)(row + k0 + 16 + half * 8);
  return __builtin_shufflevector(a, b, 0, 1, 2, 3, 4, 5, 6, 7,
                                 8, 9, 10, 11, 12, 13, 14, 15);
}

// ---------------- kernel 0: one-shot f32 -> bf16 conversion ----------------
__global__ __launch_bounds__(256) void cvt_kernel(const float* __restrict__ src,
                                                  __bf16* __restrict__ dst, int n4) {
  int i = blockIdx.x * 256 + threadIdx.x;
  if (i < n4) {
    float4 v = ((const float4*)src)[i];
    v4bf o;
    o[0] = (__bf16)v.x; o[1] = (__bf16)v.y; o[2] = (__bf16)v.z; o[3] = (__bf16)v.w;
    ((v4bf*)dst)[i] = o;
  }
}

// ---------------- kernel 1: fused input projection (+bias, +RMSNorm, head split) ----
// Y = X @ W^T + b ; per wave: 32 rows x one head (64 cols). grid = (BL/256, H)
template <bool DO_NORM, bool STORE_T>
__global__ __launch_bounds__(256) void proj_kernel(
    const __bf16* __restrict__ X,     // [BL, 1024] bf16 (preconverted)
    const __bf16* __restrict__ W,     // [1024, 1024] bf16, torch-style [out,in]
    const float* __restrict__ bias,   // [1024]
    const float* __restrict__ normw,  // [64] (ignored if !DO_NORM)
    __bf16* __restrict__ out)         // [B,H,L,64] or [B,H,64,L] if STORE_T
{
  const int wave = threadIdx.x >> 5;
  const int lane = threadIdx.x & 31;
  const int nn   = lane & 15;
  const int half = lane >> 4;
  const int h    = blockIdx.y;
  const int row0 = blockIdx.x * 256 + wave * 32;

  const __bf16* arow0 = X + (size_t)(row0 + nn) * kDIM;
  const __bf16* arow1 = X + (size_t)(row0 + 16 + nn) * kDIM;

  v8f acc[2][4] = {};
  for (int k0 = 0; k0 < kDIM; k0 += 32) {
    v16bf a0 = load_frag_bf16(arow0, k0, half);
    v16bf a1 = load_frag_bf16(arow1, k0, half);
#pragma unroll
    for (int j = 0; j < 4; ++j) {
      v16bf bf = load_frag_bf16(W + (size_t)(h * 64 + j * 16 + nn) * kDIM, k0, half);
      acc[0][j] = wmma_bf16(a0, bf, acc[0][j]);
      acc[1][j] = wmma_bf16(a1, bf, acc[1][j]);
    }
  }

#pragma unroll
  for (int mt = 0; mt < 2; ++mt) {
#pragma unroll
    for (int j = 0; j < 4; ++j) {
      float bj = bias[h * 64 + j * 16 + nn];
#pragma unroll
      for (int r = 0; r < 8; ++r) acc[mt][j][r] += bj;
    }

    if (DO_NORM) {
      float ss[8];
#pragma unroll
      for (int r = 0; r < 8; ++r) {
        float s = 0.f;
#pragma unroll
        for (int j = 0; j < 4; ++j) s += acc[mt][j][r] * acc[mt][j][r];
        ss[r] = s;
      }
#pragma unroll
      for (int off = 1; off < 16; off <<= 1)
#pragma unroll
        for (int r = 0; r < 8; ++r) ss[r] += __shfl_xor(ss[r], off);
#pragma unroll
      for (int r = 0; r < 8; ++r) ss[r] = rsqrtf(ss[r] * (1.f / 64.f) + kEPS);
#pragma unroll
      for (int j = 0; j < 4; ++j) {
        float wn = normw[j * 16 + nn];
#pragma unroll
        for (int r = 0; r < 8; ++r) acc[mt][j][r] = acc[mt][j][r] * ss[r] * wn;
      }
    }

#pragma unroll
    for (int j = 0; j < 4; ++j) {
#pragma unroll
      for (int r = 0; r < 8; ++r) {
        int mg = row0 + mt * 16 + r + 8 * half;
        int bb = mg >> 11;            // / kL
        int s  = mg & (kL - 1);
        int d  = j * 16 + nn;
        size_t off;
        if (STORE_T)
          off = ((((size_t)bb * kH + h) * kHD) + d) * kL + s;   // V^T per head
        else
          off = ((((size_t)bb * kH + h) * kL) + s) * kHD + d;   // [B,H,L,64]
        out[off] = (__bf16)acc[mt][j][r];
      }
    }
  }
}

// ---------------- kernel 2: flash attention per (b,h) ----------------
// grid = (L/128, B*H); each wave owns 16 q-rows, streams K/V in 64-wide tiles.
__global__ __launch_bounds__(256) void attn_kernel(
    const __bf16* __restrict__ Qh,   // [B,H,L,64]
    const __bf16* __restrict__ Kh,   // [B,H,L,64]
    const __bf16* __restrict__ Vt,   // [B,H,64,L]
    const int*    __restrict__ amask,// [B,L]
    __bf16* __restrict__ Oh)         // [B,L,1024] head-merged
{
  __shared__ __bf16 plds[8 * 16 * 64];   // 2 KB per wave for P transpose
  const int wave = threadIdx.x >> 5;
  const int lane = threadIdx.x & 31;
  const int nn   = lane & 15;
  const int half = lane >> 4;
  const int bh   = blockIdx.y;
  const int b    = bh / kH;
  const int h    = bh % kH;
  const int q0   = blockIdx.x * 128 + wave * 16;

  const __bf16* qbase = Qh + (((size_t)b * kH + h) * kL + q0) * kHD;
  const __bf16* kbase = Kh + (((size_t)b * kH + h) * kL) * kHD;
  const __bf16* vbase = Vt + (((size_t)b * kH + h) * kHD) * kL;
  const int*    mb    = amask + (size_t)b * kL;
  __bf16* pw = plds + wave * 16 * 64;

  // Q fragments (held for the whole loop)
  v16bf aq0 = load_frag_bf16(qbase + (size_t)nn * kHD, 0, half);
  v16bf aq1 = load_frag_bf16(qbase + (size_t)nn * kHD, 32, half);

  v8f o[4] = {};
  float mrun[8], lrun[8];
#pragma unroll
  for (int r = 0; r < 8; ++r) { mrun[r] = -1e30f; lrun[r] = 0.f; }

  for (int kt = 0; kt < kL; kt += 64) {
    // ---- S = Q K^T over this 64-col tile ----
    v8f s[4] = {};
#pragma unroll
    for (int j = 0; j < 4; ++j) {
      const __bf16* krow = kbase + (size_t)(kt + j * 16 + nn) * kHD;
      v16bf bk0 = load_frag_bf16(krow, 0, half);
      v16bf bk1 = load_frag_bf16(krow, 32, half);
      s[j] = wmma_bf16(aq0, bk0, s[j]);
      s[j] = wmma_bf16(aq1, bk1, s[j]);
    }
    // ---- scale + mask ----
#pragma unroll
    for (int j = 0; j < 4; ++j) {
      float mk = (mb[kt + j * 16 + nn] == 0) ? -1e30f : 0.f;
#pragma unroll
      for (int r = 0; r < 8; ++r) s[j][r] = s[j][r] * kSCALE + mk;
    }
    // ---- online softmax ----
    float rm[8];
#pragma unroll
    for (int r = 0; r < 8; ++r) {
      float v = s[0][r];
#pragma unroll
      for (int j = 1; j < 4; ++j) v = fmaxf(v, s[j][r]);
      rm[r] = v;
    }
#pragma unroll
    for (int off = 1; off < 16; off <<= 1)
#pragma unroll
      for (int r = 0; r < 8; ++r) rm[r] = fmaxf(rm[r], __shfl_xor(rm[r], off));

    float alpha[8], rs[8];
#pragma unroll
    for (int r = 0; r < 8; ++r) {
      float mn = fmaxf(mrun[r], rm[r]);
      alpha[r] = __expf(mrun[r] - mn);
      mrun[r]  = mn;
      rs[r]    = 0.f;
    }
#pragma unroll
    for (int j = 0; j < 4; ++j)
#pragma unroll
      for (int r = 0; r < 8; ++r) {
        float p = __expf(s[j][r] - mrun[r]);
        s[j][r] = p;
        rs[r] += p;
      }
#pragma unroll
    for (int off = 1; off < 16; off <<= 1)
#pragma unroll
      for (int r = 0; r < 8; ++r) rs[r] += __shfl_xor(rs[r], off);
#pragma unroll
    for (int r = 0; r < 8; ++r) lrun[r] = lrun[r] * alpha[r] + rs[r];
#pragma unroll
    for (int j = 0; j < 4; ++j)
#pragma unroll
      for (int r = 0; r < 8; ++r) o[j][r] *= alpha[r];

    // ---- P: accumulator layout -> A-fragment layout via per-wave LDS ----
#pragma unroll
    for (int j = 0; j < 4; ++j)
#pragma unroll
      for (int r = 0; r < 8; ++r)
        pw[(r + 8 * half) * 64 + j * 16 + nn] = (__bf16)s[j][r];
    asm volatile("s_wait_dscnt 0" ::: "memory");  // wave-internal LDS RAW

    v16bf ap0 = load_frag_bf16(pw + nn * 64, 0, half);
    v16bf ap1 = load_frag_bf16(pw + nn * 64, 32, half);

    // ---- O += P V (V pre-transposed: K-contiguous) ----
#pragma unroll
    for (int j = 0; j < 4; ++j) {
      const __bf16* vrow = vbase + (size_t)(j * 16 + nn) * kL + kt;
      v16bf bv0 = load_frag_bf16(vrow, 0, half);
      v16bf bv1 = load_frag_bf16(vrow, 32, half);
      o[j] = wmma_bf16(ap0, bv0, o[j]);
      o[j] = wmma_bf16(ap1, bv1, o[j]);
    }
  }

  // ---- normalize + store merged-head bf16 ----
  float inv[8];
#pragma unroll
  for (int r = 0; r < 8; ++r) inv[r] = (lrun[r] > 0.f) ? (1.f / lrun[r]) : 0.f;
#pragma unroll
  for (int j = 0; j < 4; ++j)
#pragma unroll
    for (int r = 0; r < 8; ++r) {
      int s = q0 + r + 8 * half;
      size_t off = ((size_t)b * kL + s) * kDIM + h * 64 + j * 16 + nn;
      Oh[off] = (__bf16)(o[j][r] * inv[r]);
    }
}

// ---------------- kernel 3: output projection (bf16 A/B, f32 out) ----------------
// grid = (BL/256, DIM/64); per wave: 32 rows x 64 cols
__global__ __launch_bounds__(256) void oproj_kernel(
    const __bf16* __restrict__ Oh,  // [BL, 1024] bf16
    const __bf16* __restrict__ Wo,  // [1024, 1024] bf16 (preconverted)
    const float*  __restrict__ bo,  // [1024]
    float* __restrict__ Y)          // [BL, 1024] f32
{
  const int wave = threadIdx.x >> 5;
  const int lane = threadIdx.x & 31;
  const int nn   = lane & 15;
  const int half = lane >> 4;
  const int nc0  = blockIdx.y * 64;
  const int row0 = blockIdx.x * 256 + wave * 32;

  const __bf16* arow0 = Oh + (size_t)(row0 + nn) * kDIM;
  const __bf16* arow1 = Oh + (size_t)(row0 + 16 + nn) * kDIM;

  v8f acc[2][4] = {};
  for (int k0 = 0; k0 < kDIM; k0 += 32) {
    v16bf a0 = load_frag_bf16(arow0, k0, half);
    v16bf a1 = load_frag_bf16(arow1, k0, half);
#pragma unroll
    for (int j = 0; j < 4; ++j) {
      v16bf bf = load_frag_bf16(Wo + (size_t)(nc0 + j * 16 + nn) * kDIM, k0, half);
      acc[0][j] = wmma_bf16(a0, bf, acc[0][j]);
      acc[1][j] = wmma_bf16(a1, bf, acc[1][j]);
    }
  }
#pragma unroll
  for (int mt = 0; mt < 2; ++mt)
#pragma unroll
    for (int j = 0; j < 4; ++j) {
      float bj = bo[nc0 + j * 16 + nn];
#pragma unroll
      for (int r = 0; r < 8; ++r) {
        int mg = row0 + mt * 16 + r + 8 * half;
        Y[(size_t)mg * kDIM + nc0 + j * 16 + nn] = acc[mt][j][r] + bj;
      }
    }
}

// ---------------- launch ----------------
extern "C" void kernel_launch(void* const* d_in, const int* in_sizes, int n_in,
                              void* d_out, int out_size, void* d_ws, size_t ws_size,
                              hipStream_t stream) {
  const float* query = (const float*)d_in[0];
  const float* key_  = (const float*)d_in[1];
  const float* value = (const float*)d_in[2];
  const int*   amask = (const int*)  d_in[3];
  const float* Wq = (const float*)d_in[4];  const float* bq = (const float*)d_in[5];
  const float* Wk = (const float*)d_in[6];  const float* bk = (const float*)d_in[7];
  const float* Wv = (const float*)d_in[8];  const float* bv = (const float*)d_in[9];
  const float* Wo = (const float*)d_in[10]; const float* bo = (const float*)d_in[11];
  const float* qnw = (const float*)d_in[12];
  const float* knw = (const float*)d_in[13];

  const size_t actElems = (size_t)kBL * kDIM;            // 4M elems
  const size_t wElems   = (size_t)kDIM * kDIM;           // 1M elems

  __bf16* Qh  = (__bf16*)d_ws;           // [B,H,L,64]
  __bf16* Kh  = Qh  + actElems;
  __bf16* Vt  = Kh  + actElems;          // [B,H,64,L]
  __bf16* Oh  = Vt  + actElems;          // [B,L,1024]
  __bf16* Xq  = Oh  + actElems;          // bf16 activations
  __bf16* Xk  = Xq  + actElems;
  __bf16* Xv  = Xk  + actElems;
  __bf16* Wqb = Xv  + actElems;          // bf16 weights
  __bf16* Wkb = Wqb + wElems;
  __bf16* Wvb = Wkb + wElems;
  __bf16* Wob = Wvb + wElems;

  dim3 blk(256);

  // one-shot f32 -> bf16 conversions (bandwidth bound, ~4 us total)
  {
    int na4 = (int)(actElems / 4), nw4 = (int)(wElems / 4);
    dim3 ga((na4 + 255) / 256), gw((nw4 + 255) / 256);
    cvt_kernel<<<ga, blk, 0, stream>>>(query, Xq, na4);
    cvt_kernel<<<ga, blk, 0, stream>>>(key_,  Xk, na4);
    cvt_kernel<<<ga, blk, 0, stream>>>(value, Xv, na4);
    cvt_kernel<<<gw, blk, 0, stream>>>(Wq, Wqb, nw4);
    cvt_kernel<<<gw, blk, 0, stream>>>(Wk, Wkb, nw4);
    cvt_kernel<<<gw, blk, 0, stream>>>(Wv, Wvb, nw4);
    cvt_kernel<<<gw, blk, 0, stream>>>(Wo, Wob, nw4);
  }

  dim3 g1(kBL / 256, kH);
  proj_kernel<true,  false><<<g1, blk, 0, stream>>>(Xq, Wqb, bq, qnw, Qh);
  proj_kernel<true,  false><<<g1, blk, 0, stream>>>(Xk, Wkb, bk, knw, Kh);
  proj_kernel<false, true ><<<g1, blk, 0, stream>>>(Xv, Wvb, bv, nullptr, Vt);

  dim3 g2(kL / 128, kB * kH);
  attn_kernel<<<g2, blk, 0, stream>>>(Qh, Kh, Vt, amask, Oh);

  dim3 g3(kBL / 256, kDIM / 64);
  oproj_kernel<<<g3, blk, 0, stream>>>(Oh, Wob, bo, (float*)d_out);
}
